// Transolver_32762010533916
// MI455X (gfx1250) — matrix operand order repaired and checked by reading, not compiled, not run on hardware
//
#include <hip/hip_runtime.h>
#include <math.h>

// ---------------------------------------------------------------------------
// Types for CDNA5 (gfx1250) WMMA, wave32
// ---------------------------------------------------------------------------
typedef __bf16 bf16_t;
typedef __attribute__((ext_vector_type(16))) __bf16 v16bf_t;
typedef __attribute__((ext_vector_type(8)))  __bf16 v8bf_t;
typedef __attribute__((ext_vector_type(8)))  float  v8f_t;

constexpr int NPTS  = 65536;
constexpr int HID   = 256;
constexpr int HEADS = 8;
constexpr int DH    = 32;
constexpr int GSL   = 32;

__device__ __forceinline__ float gelu_f(float x) {
    float x3 = x * x * x;
    return 0.5f * x * (1.f + tanhf(0.7978845608028654f * (x + 0.044715f * x3)));
}

// ---------------------------------------------------------------------------
// Weight packing: W (K x M, row-major f32) -> bf16 fragments laid out so each
// lane's 16 B-matrix elements for one v_wmma_f32_16x16x32_bf16 are contiguous.
// Layout: [nt][kc][lane 0..31][e 0..15], element = W[kc*32 + 16*(lane>>4) + e]
//                                               [nt*16 + (lane&15)]
// ---------------------------------------------------------------------------
__global__ __launch_bounds__(256) void pack_b_k(const float* __restrict__ W,
                                                bf16_t* __restrict__ out,
                                                int K, int M) {
    int idx = blockIdx.x * 256 + threadIdx.x;
    if (idx >= K * M) return;
    int e    = idx & 15;
    int lane = (idx >> 4) & 31;
    int rem  = idx >> 9;
    int kch  = K >> 5;
    int kc   = rem % kch;
    int nt   = rem / kch;
    int n = nt * 16 + (lane & 15);
    int k = kc * 32 + 16 * (lane >> 4) + e;
    out[idx] = (bf16_t)W[k * M + n];
}

// ---------------------------------------------------------------------------
// WMMA GEMM: C = epilogue(A @ B + bias [+extra] [+resid])
// A: (NPTS x K) bf16 row-major, Bp: packed fragments. Block = 8 waves, each
// wave computes 16 rows x (MTILES*16) cols. Per-head mode via blockIdx.z.
// ---------------------------------------------------------------------------
template<int MTILES>
__global__ __launch_bounds__(256) void gemm_wmma_k(
    const bf16_t* __restrict__ A, const bf16_t* __restrict__ Bp,
    const float* __restrict__ bias, const float* __restrict__ extra,
    const bf16_t* __restrict__ resid, bf16_t* __restrict__ Cout,
    int K, int ldc, size_t aHeadStride, size_t bHeadStride,
    int colHeadStride, int act)
{
    constexpr int ROWS = 128;
    constexpr int AST  = 40;                       // padded LDS stride (halfwords)
    __shared__ __align__(16) bf16_t As[ROWS * AST];

    const int h = blockIdx.z;
    A  += (size_t)h * aHeadStride;
    Bp += (size_t)h * bHeadStride;
    const int colOff = h * colHeadStride;

    const int tid  = threadIdx.x;
    const int wave = tid >> 5;
    const int lane = tid & 31;
    const size_t row0 = (size_t)blockIdx.x * ROWS;
    const int kchunks = K >> 5;

    v8f_t acc[MTILES] = {};

    for (int kc = 0; kc < kchunks; ++kc) {
        __syncthreads();
        // cooperative A-chunk stage: 128 rows x 32 halfwords (coalesced 16B loads)
        #pragma unroll
        for (int j = 0; j < 2; ++j) {
            int s  = tid * 2 + j;
            int r  = s >> 2;
            int cs = s & 3;
            const v8bf_t* src =
                (const v8bf_t*)(A + (row0 + r) * (size_t)K + (size_t)kc * 32 + cs * 8);
            *(v8bf_t*)(As + r * AST + cs * 8) = *src;
        }
        if (kc + 1 < kchunks) {   // CDNA5 global_prefetch_b8 of next chunk
            int r = (tid * 2) >> 2, cs = (tid * 2) & 3;
            __builtin_prefetch(A + (row0 + r) * (size_t)K + (size_t)(kc + 1) * 32 + cs * 8, 0, 1);
        }
        __syncthreads();

        // A fragment: contiguous k = 16*hh + e (matches packed B)
        const int m  = lane & 15;
        const int hh = lane >> 4;
        const bf16_t* arow = As + (wave * 16 + m) * AST;
        union { v16bf_t v; v8bf_t p[2]; } af;
        af.p[0] = *(const v8bf_t*)(arow + 16 * hh);
        af.p[1] = *(const v8bf_t*)(arow + 16 * hh + 8);

        #pragma unroll
        for (int nt = 0; nt < MTILES; ++nt) {
            const v16bf_t bfrag =
                *(const v16bf_t*)(Bp + ((size_t)(nt * kchunks + kc) * 32 + lane) * 16);
            acc[nt] = __builtin_amdgcn_wmma_f32_16x16x32_bf16(
                false, af.v, false, bfrag, (short)0, acc[nt], false, false);
        }
    }

    // epilogue: C/D layout row = r + 8*(lane>>4), col = lane&15
    const int m0 = 8 * (lane >> 4);
    const int nl = lane & 15;
    #pragma unroll
    for (int nt = 0; nt < MTILES; ++nt) {
        const int col = nt * 16 + nl;
        float bv = bias ? bias[col] : 0.f;
        if (extra) bv += extra[col];
        const int gcol = colOff + col;
        #pragma unroll
        for (int r = 0; r < 8; ++r) {
            size_t grow = row0 + wave * 16 + m0 + r;
            float v = acc[nt][r] + bv;
            if (act) v = gelu_f(v);
            if (resid) v += (float)resid[grow * (size_t)ldc + gcol];
            Cout[grow * (size_t)ldc + gcol] = (bf16_t)v;
        }
    }
}

// ---------------------------------------------------------------------------
// Pre-MLP first stage: h1 = gelu(x @ W1 + b1), K=7 (too skinny for WMMA)
// ---------------------------------------------------------------------------
__global__ __launch_bounds__(256) void pre_hidden_k(const float* __restrict__ x,
                                                    const float* __restrict__ W1,
                                                    const float* __restrict__ b1,
                                                    bf16_t* __restrict__ h1) {
    int idx = blockIdx.x * 256 + threadIdx.x;     // < NPTS*512
    int n = idx >> 9, j = idx & 511;
    float s = b1[j];
    #pragma unroll
    for (int k = 0; k < 7; ++k) s += x[n * 7 + k] * W1[k * 512 + j];
    h1[idx] = (bf16_t)gelu_f(s);
}

// ---------------------------------------------------------------------------
// LayerNorm over 256 cols: one wave32 per row, 8 cols per lane
// ---------------------------------------------------------------------------
__global__ __launch_bounds__(256) void layernorm_k(const bf16_t* __restrict__ x,
                                                   const float* __restrict__ w,
                                                   const float* __restrict__ b,
                                                   bf16_t* __restrict__ y) {
    size_t row = (size_t)blockIdx.x * 8 + (threadIdx.x >> 5);
    int lane = threadIdx.x & 31;
    const bf16_t* xr = x + row * HID + lane * 8;
    float v[8];
    float s = 0.f;
    #pragma unroll
    for (int i = 0; i < 8; ++i) { v[i] = (float)xr[i]; s += v[i]; }
    #pragma unroll
    for (int m = 16; m >= 1; m >>= 1) s += __shfl_xor(s, m, 32);
    float mu = s * (1.f / 256.f);
    float vs = 0.f;
    #pragma unroll
    for (int i = 0; i < 8; ++i) { float d = v[i] - mu; vs += d * d; }
    #pragma unroll
    for (int m = 16; m >= 1; m >>= 1) vs += __shfl_xor(vs, m, 32);
    float inv = rsqrtf(vs * (1.f / 256.f) + 1e-5f);
    int c0 = lane * 8;
    bf16_t* yr = y + row * HID + c0;
    #pragma unroll
    for (int i = 0; i < 8; ++i)
        yr[i] = (bf16_t)((v[i] - mu) * inv * w[c0 + i] + b[c0 + i]);
}

// ---------------------------------------------------------------------------
// Slice assignment: logits = x_mid_head @ Wslice + b, /temp[h], softmax over G.
// Writes sw per head (h-major) + accumulates snorm[h][g] via LDS + atomics.
// One thread per (point, head); 32 points x 8 heads per block.
// ---------------------------------------------------------------------------
__global__ __launch_bounds__(256) void slice_softmax_k(
    const bf16_t* __restrict__ xmid, const float* __restrict__ Wslice,
    const float* __restrict__ bslice, const float* __restrict__ temp,
    bf16_t* __restrict__ sw, float* __restrict__ snorm)
{
    __shared__ float Ws[32][33];
    __shared__ float snp[256];
    int tid = threadIdx.x;
    for (int q = tid; q < 1024; q += 256) Ws[q >> 5][q & 31] = Wslice[q];
    snp[tid] = 0.f;
    __syncthreads();

    int p = tid >> 3, h = tid & 7;
    size_t n = (size_t)blockIdx.x * 32 + p;
    const bf16_t* xr = xmid + n * HID + h * DH;
    float xv[32];
    #pragma unroll
    for (int d = 0; d < 32; ++d) xv[d] = (float)xr[d];

    float it = 1.f / temp[h];
    float lg[32];
    float mx = -1e30f;
    #pragma unroll
    for (int g = 0; g < 32; ++g) {
        float s = bslice[g];
        #pragma unroll
        for (int d = 0; d < 32; ++d) s += xv[d] * Ws[d][g];
        s *= it;
        lg[g] = s;
        mx = fmaxf(mx, s);
    }
    float den = 0.f;
    #pragma unroll
    for (int g = 0; g < 32; ++g) { lg[g] = __expf(lg[g] - mx); den += lg[g]; }
    float inv = 1.f / den;

    bf16_t* swr = sw + (size_t)h * NPTS * GSL + n * GSL;
    #pragma unroll
    for (int g = 0; g < 32; ++g) {
        float v = lg[g] * inv;
        swr[g] = (bf16_t)v;
        atomicAdd(&snp[h * 32 + g], v);
    }
    __syncthreads();
    atomicAdd(&snorm[tid], snp[tid]);
}

// ---------------------------------------------------------------------------
// Token reduction: tok[h][g][d] += sum_n sw[h][n][g] * fx_mid[n][h*32+d]
// grid = (NPTS/256, HEADS); 256 threads: g = t&31, point-subset = t>>5
// ---------------------------------------------------------------------------
__global__ __launch_bounds__(256) void tok_reduce_k(const bf16_t* __restrict__ fxmid,
                                                    const bf16_t* __restrict__ sw,
                                                    float* __restrict__ tok) {
    __shared__ float fxs[256][32];
    __shared__ float sws[256][32];
    __shared__ float red[32][33];
    int tid = threadIdx.x;
    int h = blockIdx.y;
    size_t n0 = (size_t)blockIdx.x * 256;
    {
        const bf16_t* fr = fxmid + (n0 + tid) * HID + h * DH;
        const bf16_t* sr = sw + (size_t)h * NPTS * GSL + (n0 + tid) * GSL;
        #pragma unroll
        for (int d = 0; d < 32; ++d) { fxs[tid][d] = (float)fr[d]; sws[tid][d] = (float)sr[d]; }
    }
    for (int q = tid; q < 32 * 33; q += 256) ((float*)red)[q] = 0.f;
    __syncthreads();

    int g = tid & 31, sub = tid >> 5;
    float acc[32] = {};
    #pragma unroll 4
    for (int pp = 0; pp < 32; ++pp) {
        int p = sub * 32 + pp;
        float s = sws[p][g];
        #pragma unroll
        for (int d = 0; d < 32; ++d) acc[d] += s * fxs[p][d];
    }
    #pragma unroll
    for (int d = 0; d < 32; ++d) atomicAdd(&red[g][d], acc[d]);
    __syncthreads();
    for (int q = tid; q < 1024; q += 256)
        atomicAdd(&tok[h * 1024 + q], red[q >> 5][q & 31]);
}

// ---------------------------------------------------------------------------
// Tiny per-head token attention on 32x32 tokens (one wave per head)
// ---------------------------------------------------------------------------
__global__ void token_attn_k(const float* __restrict__ tokacc,
                             const float* __restrict__ snorm,
                             const float* __restrict__ Wq,
                             const float* __restrict__ Wk,
                             const float* __restrict__ Wv,
                             float* __restrict__ outTok) {
    __shared__ float tk[32][33], kk[32][33], vv[32][33];
    int h = blockIdx.x, g = threadIdx.x;
    float inv = 1.f / (snorm[h * 32 + g] + 1e-5f);
    #pragma unroll
    for (int c = 0; c < 32; ++c) tk[g][c] = tokacc[h * 1024 + g * 32 + c] * inv;
    __syncthreads();
    float q[32];
    #pragma unroll
    for (int d = 0; d < 32; ++d) {
        float sq = 0.f, sk = 0.f, sv = 0.f;
        #pragma unroll
        for (int c = 0; c < 32; ++c) {
            float t = tk[g][c];
            sq += t * Wq[c * 32 + d];
            sk += t * Wk[c * 32 + d];
            sv += t * Wv[c * 32 + d];
        }
        q[d] = sq; kk[g][d] = sk; vv[g][d] = sv;
    }
    __syncthreads();
    float att[32];
    float mx = -1e30f;
    #pragma unroll
    for (int j = 0; j < 32; ++j) {
        float s = 0.f;
        #pragma unroll
        for (int d = 0; d < 32; ++d) s += q[d] * kk[j][d];
        s *= 0.17677669529663687f;   // 32^-0.5
        att[j] = s;
        mx = fmaxf(mx, s);
    }
    float den = 0.f;
    #pragma unroll
    for (int j = 0; j < 32; ++j) { att[j] = __expf(att[j] - mx); den += att[j]; }
    float rin = 1.f / den;
    #pragma unroll
    for (int d = 0; d < 32; ++d) {
        float o = 0.f;
        #pragma unroll
        for (int j = 0; j < 32; ++j) o += att[j] * vv[j][d];
        outTok[h * 1024 + g * 32 + d] = o * rin;
    }
}

// ---------------------------------------------------------------------------
// Final head: out = g2 @ m2_W2 + b   (256 -> 4, f32 output)
// ---------------------------------------------------------------------------
__global__ __launch_bounds__(256) void final_out_k(const bf16_t* __restrict__ g2,
                                                   const float* __restrict__ W2,
                                                   const float* __restrict__ b2,
                                                   float* __restrict__ out) {
    __shared__ float Ws[1024];
    int tid = threadIdx.x;
    for (int q = tid; q < 1024; q += 256) Ws[q] = W2[q];
    __syncthreads();
    size_t n = (size_t)blockIdx.x * 256 + tid;
    float a0 = b2[0], a1 = b2[1], a2 = b2[2], a3 = b2[3];
    const bf16_t* r = g2 + n * HID;
    #pragma unroll 8
    for (int k = 0; k < 256; ++k) {
        float xv = (float)r[k];
        a0 += xv * Ws[k * 4 + 0];
        a1 += xv * Ws[k * 4 + 1];
        a2 += xv * Ws[k * 4 + 2];
        a3 += xv * Ws[k * 4 + 3];
    }
    out[n * 4 + 0] = a0; out[n * 4 + 1] = a1;
    out[n * 4 + 2] = a2; out[n * 4 + 3] = a3;
}

// ---------------------------------------------------------------------------
// Host orchestration (all on `stream`, deterministic, graph-capture safe)
// ---------------------------------------------------------------------------
extern "C" void kernel_launch(void* const* d_in, const int* in_sizes, int n_in,
                              void* d_out, int out_size, void* d_ws, size_t ws_size,
                              hipStream_t stream) {
    (void)in_sizes; (void)n_in; (void)out_size; (void)ws_size;

    const float* x           = (const float*)d_in[0];
    const float* preW1       = (const float*)d_in[1];
    const float* preb1       = (const float*)d_in[2];
    const float* preW2       = (const float*)d_in[3];
    const float* preb2       = (const float*)d_in[4];
    const float* placeholder = (const float*)d_in[5];
    auto BP = [&](int i, int j) -> const float* { return (const float*)d_in[6 + i * 20 + j]; };

    char* base = (char*)d_ws;
    size_t off = 0;
    auto alloc = [&](size_t bytes) -> char* {
        char* p = base + off;
        off += (bytes + 255) & ~(size_t)255;
        return p;
    };

    // packed bf16 weights: preW2 (512x256) + 25 x (256x256) + m2W1 (256x256)
    bf16_t* packAll   = (bf16_t*)alloc(sizeof(bf16_t) * 1835008);
    bf16_t* packPreW2 = packAll;
    auto packL = [&](int l, int j) { return packAll + 131072 + (size_t)(l * 5 + j) * 65536; };
    bf16_t* packM2W1 = packAll + 131072 + (size_t)25 * 65536;

    bf16_t* h1   = (bf16_t*)alloc(sizeof(bf16_t) * (size_t)NPTS * 512);
    bf16_t* fx   = (bf16_t*)alloc(sizeof(bf16_t) * (size_t)NPTS * HID);
    bf16_t* xn   = (bf16_t*)alloc(sizeof(bf16_t) * (size_t)NPTS * HID);
    bf16_t* bufA = (bf16_t*)alloc(sizeof(bf16_t) * (size_t)NPTS * HID);  // fx_mid / mlp hidden
    bf16_t* bufB = (bf16_t*)alloc(sizeof(bf16_t) * (size_t)NPTS * HID);  // x_mid
    bf16_t* attn = (bf16_t*)alloc(sizeof(bf16_t) * (size_t)NPTS * HID);  // broadcast out
    bf16_t* swb  = (bf16_t*)alloc(sizeof(bf16_t) * (size_t)HEADS * NPTS * GSL);
    float*  tokacc = (float*)alloc(sizeof(float) * (8192 + 256));
    float*  snorm  = tokacc + 8192;
    float*  outTok = (float*)alloc(sizeof(float) * 8192);
    bf16_t* packTok = (bf16_t*)alloc(sizeof(bf16_t) * 8192);

    // ---- pack all static weights to WMMA B-fragment layout (bf16) ----
    pack_b_k<<<512, 256, 0, stream>>>(preW2, packPreW2, 512, 256);
    const int widx[5] = {2, 4, 12, 16, 18};   // Wfx, Wx, Wo, m_W1, m_W2
    for (int l = 0; l < 5; ++l)
        for (int j = 0; j < 5; ++j)
            pack_b_k<<<256, 256, 0, stream>>>(BP(l, widx[j]), packL(l, j), 256, 256);
    pack_b_k<<<256, 256, 0, stream>>>(BP(4, 22), packM2W1, 256, 256);

    // ---- preprocess: fx = gelu(x@W1+b1)@W2 + b2 + placeholder ----
    pre_hidden_k<<<(NPTS * 512) / 256, 256, 0, stream>>>(x, preW1, preb1, h1);
    gemm_wmma_k<16><<<dim3(NPTS / 128, 1, 1), 256, 0, stream>>>(
        h1, packPreW2, preb2, placeholder, nullptr, fx, 512, HID, 0, 0, 0, 0);

    // ---- 5 transformer blocks ----
    for (int l = 0; l < 5; ++l) {
        layernorm_k<<<NPTS / 8, 256, 0, stream>>>(fx, BP(l, 0), BP(l, 1), xn);
        gemm_wmma_k<16><<<dim3(NPTS / 128, 1, 1), 256, 0, stream>>>(
            xn, packL(l, 0), BP(l, 3), nullptr, nullptr, bufA, 256, HID, 0, 0, 0, 0);
        gemm_wmma_k<16><<<dim3(NPTS / 128, 1, 1), 256, 0, stream>>>(
            xn, packL(l, 1), BP(l, 5), nullptr, nullptr, bufB, 256, HID, 0, 0, 0, 0);

        hipMemsetAsync(tokacc, 0, sizeof(float) * (8192 + 256), stream);
        slice_softmax_k<<<NPTS / 32, 256, 0, stream>>>(
            bufB, BP(l, 6), BP(l, 7), BP(l, 8), swb, snorm);
        tok_reduce_k<<<dim3(NPTS / 256, HEADS, 1), 256, 0, stream>>>(bufA, swb, tokacc);
        token_attn_k<<<HEADS, 32, 0, stream>>>(
            tokacc, snorm, BP(l, 9), BP(l, 10), BP(l, 11), outTok);

        for (int h = 0; h < HEADS; ++h)  // pack out_tok (32x32) per head for WMMA
            pack_b_k<<<4, 256, 0, stream>>>(outTok + h * 1024, packTok + h * 1024, 32, 32);
        // broadcast: attn[:, h*32:(h+1)*32] = sw_h @ out_tok_h  (K=32 WMMA, head=grid.z)
        gemm_wmma_k<2><<<dim3(NPTS / 128, 1, HEADS), 256, 0, stream>>>(
            swb, packTok, nullptr, nullptr, nullptr, attn, 32, HID,
            (size_t)NPTS * GSL, 1024, 32, 0);

        // Wo projection + residual
        gemm_wmma_k<16><<<dim3(NPTS / 128, 1, 1), 256, 0, stream>>>(
            attn, packL(l, 2), BP(l, 13), nullptr, fx, fx, 256, HID, 0, 0, 0, 0);

        // MLP
        layernorm_k<<<NPTS / 8, 256, 0, stream>>>(fx, BP(l, 14), BP(l, 15), xn);
        gemm_wmma_k<16><<<dim3(NPTS / 128, 1, 1), 256, 0, stream>>>(
            xn, packL(l, 3), BP(l, 17), nullptr, nullptr, bufA, 256, HID, 0, 0, 0, 1);
        gemm_wmma_k<16><<<dim3(NPTS / 128, 1, 1), 256, 0, stream>>>(
            bufA, packL(l, 4), BP(l, 19), nullptr, fx, fx, 256, HID, 0, 0, 0, 0);
    }

    // ---- final head ----
    layernorm_k<<<NPTS / 8, 256, 0, stream>>>(fx, BP(4, 20), BP(4, 21), xn);
    gemm_wmma_k<16><<<dim3(NPTS / 128, 1, 1), 256, 0, stream>>>(
        xn, packM2W1, BP(4, 23), nullptr, nullptr, bufA, 256, HID, 0, 0, 0, 1);
    final_out_k<<<NPTS / 256, 256, 0, stream>>>(bufA, BP(4, 24), BP(4, 25), (float*)d_out);
}
